// ConverterCell_80839874445808
// MI455X (gfx1250) — matrix operand order repaired and checked by reading, not compiled, not run on hardware
//
#include <hip/hip_runtime.h>
#include <hip/hip_bf16.h>

#define BN 256
#define LAY 8
#define EPSV 1e-3f

typedef __attribute__((ext_vector_type(16))) __bf16 v16bf;
typedef __attribute__((ext_vector_type(8)))  float  v8f;

__device__ __forceinline__ __bf16 f2bf(float f) { return (__bf16)f; }

// ---- CDNA5 async global->LDS copy (ASYNCcnt-tracked) ----------------------
__device__ __forceinline__ void async_b128(unsigned lds_off, const void* gaddr)
{
  asm volatile("global_load_async_to_lds_b128 %0, %1, off"
               :: "v"(lds_off), "v"(gaddr) : "memory");
}
__device__ __forceinline__ void wait_async0()
{
#if __has_builtin(__builtin_amdgcn_s_wait_asynccnt)
  __builtin_amdgcn_s_wait_asynccnt(0);
#else
  asm volatile("s_wait_asynccnt 0x0" ::: "memory");
#endif
}
__device__ __forceinline__ unsigned lds_addr(const void* p)
{
  return (unsigned)(unsigned long long)p;   // low 32 bits = group-segment offset
}

// Build the 16-bit A-fragment (contiguous K order: per lane-half h, K = h*16+0..15;
// B uses the same order) from four aligned float4 loads.
__device__ __forceinline__ v16bf pack_a(float4 x0, float4 x1, float4 x2, float4 x3)
{
  v16bf a;
  a[0]  = (__bf16)x0.x; a[1]  = (__bf16)x0.y; a[2]  = (__bf16)x0.z; a[3]  = (__bf16)x0.w;
  a[4]  = (__bf16)x1.x; a[5]  = (__bf16)x1.y; a[6]  = (__bf16)x1.z; a[7]  = (__bf16)x1.w;
  a[8]  = (__bf16)x2.x; a[9]  = (__bf16)x2.y; a[10] = (__bf16)x2.z; a[11] = (__bf16)x2.w;
  a[12] = (__bf16)x3.x; a[13] = (__bf16)x3.y; a[14] = (__bf16)x3.z; a[15] = (__bf16)x3.w;
  return a;
}

// ---------------------------------------------------------------------------
// WMMA GEMM: block = 128 M-rows x 64 N-cols (8 waves, one 16x64 tile each).
// The shared B panel is streamed through LDS in 32-K chunks with double-
// buffered global_load_async_to_lds_b128 (loaded once per block, not per
// wave), overlapping the next chunk's copy with the current chunk's 4 WMMAs.
// C[b*rowsC + rowOff + p, :] = A[b*rowsA + p, :] (f32 MxK) x W(bf16, N x K
// transposed) + bias.  Requires M%128==0, N%64==0, K%32==0 (true here).
// ---------------------------------------------------------------------------
__global__ void k_gemm(const float* __restrict__ A, const __bf16* __restrict__ Wt,
                       const float* __restrict__ bias, float* __restrict__ C,
                       int M, int N, int K, int rowsA, int rowsC, int rowOff)
{
  __shared__ __align__(32) __bf16 bpan[2][64 * 32];
  int nBlocks = N >> 6;
  int mBlk = blockIdx.x / nBlocks;
  int nBlk = blockIdx.x - mBlk * nBlocks;
  int tid  = threadIdx.x;
  int wave = tid >> 5, lane = tid & 31, half = lane >> 4, r = lane & 15;
  int tileM = (mBlk << 7) + (wave << 4);
  int tileN = nBlk << 6;
  const float* arow = A + (size_t)(tileM + r) * K;
  // panel loader: thread tid copies 8 bf16 (16B) of the 64x32 chunk
  int ln = tid >> 2;                    // panel row (= B column) 0..63
  int lk = (tid & 3) << 3;              // k offset within chunk
  const __bf16* gsrc = Wt + (size_t)(tileN + ln) * K + lk;
  unsigned lds0 = lds_addr(&bpan[0][ln * 32 + lk]);
  unsigned lds1 = lds_addr(&bpan[1][ln * 32 + lk]);
  async_b128(lds0, gsrc);               // prologue: chunk 0 in flight
  int nch = K >> 5;
  v8f acc[4] = {};
  for (int c = 0; c < nch; ++c) {
    wait_async0();
    __syncthreads();                    // chunk c visible to all waves
    int buf = c & 1;
    if (c + 1 < nch)                    // prefetch chunk c+1 into other buffer
      async_b128(buf ? lds0 : lds1, gsrc + ((size_t)(c + 1) << 5));
    int ka = (c << 5) + half * 16;
    v16bf a = pack_a(*(const float4*)(arow + ka),
                     *(const float4*)(arow + ka + 4),
                     *(const float4*)(arow + ka + 8),
                     *(const float4*)(arow + ka + 12));
    const __bf16* bp = bpan[buf];
    #pragma unroll
    for (int t = 0; t < 4; ++t) {
      v16bf bfr = *(const v16bf*)&bp[(t * 16 + r) * 32 + half * 16];
      acc[t] = __builtin_amdgcn_wmma_f32_16x16x32_bf16(false, a, false, bfr,
                                                       (short)0, acc[t], false, false);
    }
  }
  float bv[4];
  #pragma unroll
  for (int t = 0; t < 4; ++t) bv[t] = bias ? bias[tileN + t * 16 + r] : 0.f;
  #pragma unroll
  for (int v = 0; v < 8; ++v) {         // D layout: row = v + 8*half
    int m    = tileM + v + (half << 3);
    int bidx = m / rowsA, p = m % rowsA;
    size_t crow = ((size_t)bidx * rowsC + rowOff + p) * (size_t)N;
    #pragma unroll
    for (int t = 0; t < 4; ++t)
      C[crow + tileN + t * 16 + r] = acc[t][v] + bv[t];
  }
}

// Weight prep: [mat][K][N] f32  ->  [mat][N][K] bf16 (transposed for B frags)
__global__ void k_wt(const float* __restrict__ W, __bf16* __restrict__ Wt,
                     int K, int N, long total)
{
  long idx = (long)blockIdx.x * blockDim.x + threadIdx.x;
  if (idx >= total) return;
  long mat = idx / ((long)K * N);
  int  rem = (int)(idx - mat * (long)K * N);
  int  k   = rem / N, n = rem % N;
  Wt[mat * (long)K * N + (long)n * K + k] = f2bf(W[idx]);
}

// Split x -> last (B,64,64) and mstate (B,16,64) = state + swapped tail
__global__ void k_pre(const float* __restrict__ x, const float* __restrict__ st,
                      float* __restrict__ lastb, float* __restrict__ mst, long total)
{
  long idx = (long)blockIdx.x * blockDim.x + threadIdx.x;
  if (idx >= total) return;
  long b = idx / 5120; int r = (int)(idx % 5120);
  int j = r / 80, e = r % 80;
  float v = x[idx];
  if (e < 64) lastb[(b << 12) + (j << 6) + e] = v;
  else {
    int m = e - 64;
    mst[((b << 4) + m) * 64 + j] = st[(b << 10) + (m << 6) + j] + v;
  }
}

// im2col over the 4x4x4 spatial grid (64 channels), kernel kk, out dim nn
__global__ void k_im2col(const float* __restrict__ lastb, float* __restrict__ col,
                         int kk, int nn, long total)
{
  long idx = (long)blockIdx.x * blockDim.x + threadIdx.x;
  if (idx >= total) return;
  int Kp = kk * kk * kk * 64;
  int P  = nn * nn * nn;
  long row = idx / Kp; int e = (int)(idx % Kp);
  int b = (int)(row / P), p = (int)(row % P);
  int dz = e / (kk * kk * 64); int r1 = e % (kk * kk * 64);
  int dy = r1 / (kk * 64);     int r2 = r1 % (kk * 64);
  int dx = r2 / 64, c = r2 & 63;
  int oz = p / (nn * nn), oy = (p / nn) % nn, ox = p % nn;
  int pos = (oz + dz) * 16 + (oy + dy) * 4 + (ox + dx);
  col[idx] = lastb[((size_t)b * 64 + pos) * 64 + c];
}

// Attention core: per (batch, head) block; DK=8 so VALU dot products.
__global__ void k_attn(const float* __restrict__ q, const float* __restrict__ k,
                       const float* __restrict__ v, float* __restrict__ o, int S)
{
  __shared__ float sc[128 * 128];
  int b = blockIdx.x, h = blockIdx.y, tid = threadIdx.x;
  const float* qb = q + (size_t)b * S * 64 + h * 8;
  const float* kb = k + (size_t)b * S * 64 + h * 8;
  const float* vb = v + (size_t)b * S * 64 + h * 8;
  const float scale = 0.35355339059327373f;   // 1/sqrt(8)
  for (int idx = tid; idx < S * S; idx += 256) {
    int qi = idx / S, si = idx % S;
    float acc = 0.f;
    #pragma unroll
    for (int d = 0; d < 8; ++d) acc += qb[qi * 64 + d] * kb[si * 64 + d];
    sc[idx] = acc * scale;
  }
  __syncthreads();
  for (int qi = tid; qi < S; qi += 256) {
    float mx = -1e30f;
    for (int si = 0; si < S; ++si) mx = fmaxf(mx, sc[qi * S + si]);
    float sum = 0.f;
    for (int si = 0; si < S; ++si) { float e = __expf(sc[qi * S + si] - mx); sc[qi * S + si] = e; sum += e; }
    float inv = 1.f / sum;
    for (int si = 0; si < S; ++si) sc[qi * S + si] *= inv;
  }
  __syncthreads();
  for (int idx = tid; idx < S * 8; idx += 256) {
    int qi = idx >> 3, d = idx & 7;
    float acc = 0.f;
    for (int si = 0; si < S; ++si) acc += sc[qi * S + si] * vb[si * 64 + d];
    o[((size_t)b * S + qi) * 64 + h * 8 + d] = acc;
  }
}

// FF (scalar squeeze-excite) + residual + LayerNorm, one block per batch elem
__global__ void k_ffln(const float* __restrict__ amha, int S,
                       const float* __restrict__ ffw, const float* __restrict__ resid,
                       const float* __restrict__ g, const float* __restrict__ bvec,
                       float* __restrict__ outp)
{
  __shared__ float red[256];
  __shared__ float sval, s2val;
  int b = blockIdx.x, tid = threadIdx.x;
  const float* ab = amha + (size_t)b * S * 64;
  float p = 0.f;
  for (int idx = tid; idx < S * 64; idx += 256) p += ab[idx];
  red[tid] = p; __syncthreads();
  for (int o2 = 128; o2; o2 >>= 1) { if (tid < o2) red[tid] += red[tid + o2]; __syncthreads(); }
  if (!tid) sval = red[0];
  __syncthreads();
  float s = sval;
  p = 0.f;
  for (int idx = tid; idx < 4096; idx += 256) { float hh = s * ffw[idx]; p += hh > 0.f ? hh : 0.f; }
  red[tid] = p; __syncthreads();
  for (int o2 = 128; o2; o2 >>= 1) { if (tid < o2) red[tid] += red[tid + o2]; __syncthreads(); }
  if (!tid) s2val = red[0];
  __syncthreads();
  float s2 = s2val;
  const float* w1 = ffw + 4096;
  int grp = tid >> 6, lanej = tid & 63;
  for (int j0 = 0; j0 < 64; j0 += 4) {
    int j = j0 + grp;
    float val = s2 * w1[j * 64 + lanej] + resid[((size_t)b * 64 + j) * 64 + lanej];
    red[tid] = val; __syncthreads();
    for (int o2 = 32; o2; o2 >>= 1) { if (lanej < o2) red[tid] += red[tid + o2]; __syncthreads(); }
    float mean = red[grp << 6] * 0.015625f;
    __syncthreads();
    float dd = val - mean;
    red[tid] = dd * dd; __syncthreads();
    for (int o2 = 32; o2; o2 >>= 1) { if (lanej < o2) red[tid] += red[tid + o2]; __syncthreads(); }
    float var = red[grp << 6] * 0.015625f;
    outp[((size_t)b * 64 + j) * 64 + lanej] = dd * rsqrtf(var + EPSV) * g[lanej] + bvec[lanej];
    __syncthreads();
  }
}

// out = LN(xx + yy) * g + bb, one 64-wide block per (b, row); in-place-safe on yy
__global__ void k_addln(const float* __restrict__ xx, const float* __restrict__ yy,
                        const float* __restrict__ g, const float* __restrict__ bb,
                        float* __restrict__ o)
{
  __shared__ float red[64];
  int row = blockIdx.x, c = threadIdx.x;
  size_t base = (size_t)row * 64;
  float val = xx[base + c] + yy[base + c];
  red[c] = val; __syncthreads();
  for (int o2 = 32; o2; o2 >>= 1) { if (c < o2) red[c] += red[c + o2]; __syncthreads(); }
  float mean = red[0] * 0.015625f; __syncthreads();
  float dd = val - mean;
  red[c] = dd * dd; __syncthreads();
  for (int o2 = 32; o2; o2 >>= 1) { if (c < o2) red[c] += red[c + o2]; __syncthreads(); }
  float var = red[0] * 0.015625f;
  o[base + c] = dd * rsqrtf(var + EPSV) * g[c] + bb[c];
}

// gp[b,t,f] = d0[b,f,t]
__global__ void k_tr(const float* __restrict__ in, float* __restrict__ outp, long total)
{
  long idx = (long)blockIdx.x * blockDim.x + threadIdx.x;
  if (idx >= total) return;
  long b = idx >> 12; int r = (int)(idx & 4095);
  int t = r >> 6, f = r & 63;
  outp[idx] = in[(b << 12) + (f << 6) + t];
}

// ap[b,m,t]: m<64 -> fore[b,t,m]; else back[b,t,m-64]
__global__ void k_ap(const float* __restrict__ ysF, const float* __restrict__ ysB,
                     float* __restrict__ ap, long total)
{
  long idx = (long)blockIdx.x * blockDim.x + threadIdx.x;
  if (idx >= total) return;
  long b = idx / 8192; int r = (int)(idx % 8192);
  int m = r >> 6, t = r & 63;
  ap[idx] = (m < 64) ? ysF[(b << 12) + (t << 6) + m]
                     : ysB[(b << 12) + (t << 6) + (m - 64)];
}

// ---------------------------------------------------------------------------
// Fused persistent GRU: one block owns 16 batch rows for all 64 timesteps.
// h (f32) and the 192x64 bf16 recurrent weight live in LDS (weight brought in
// with async-to-LDS); per step waves 0..5 each compute a 16x32 tile of h @ r
// (2 WMMA accumulators sharing one A fragment), then 256 threads do the gate
// update.  grid = (B/16, 2 directions)
// ---------------------------------------------------------------------------
__global__ void k_gru(const float* __restrict__ xgF, const float* __restrict__ xgB,
                      const __bf16* __restrict__ rTF, const __bf16* __restrict__ rTB,
                      const float* __restrict__ brF, const float* __restrict__ brB,
                      const float* __restrict__ mst, int layer,
                      float* __restrict__ ysF, float* __restrict__ ysB,
                      float* __restrict__ stb)
{
  __shared__ __align__(16) float  h[16 * 64];
  __shared__ float  hg[16 * 192];
  __shared__ __align__(32) __bf16 rT[192 * 64];
  int dir = blockIdx.y;
  const float*  xg  = dir ? xgB : xgF;
  const __bf16* rTg = dir ? rTB : rTF;
  const float*  br  = dir ? brB : brF;
  float*        ys  = dir ? ysB : ysF;
  int b0 = blockIdx.x * 16;
  int tid = threadIdx.x;
  for (int base = tid * 8; base < 192 * 64; base += 256 * 8)     // async 24KB
    async_b128(lds_addr(&rT[base]), rTg + base);
  for (int idx = tid; idx < 16 * 64; idx += 256) {
    int row = idx >> 6, j = idx & 63;
    h[idx] = mst[(((size_t)(b0 + row) * 8 + layer) * 2 + dir) * 64 + j];
  }
  wait_async0();
  __syncthreads();
  int wave = tid >> 5, lane = tid & 31, half = lane >> 4, r = lane & 15;
  for (int t = 0; t < 64; ++t) {
    if (wave < 6) {                       // hg = h(16x64) @ r(64x192): 6 x 16x32
      int tn = wave * 32;
      v8f acc0 = {}, acc1 = {};
      #pragma unroll
      for (int k0 = 0; k0 < 64; k0 += 32) {
        int ka = k0 + half * 16;
        v16bf a = pack_a(*(const float4*)(&h[r * 64 + ka]),
                         *(const float4*)(&h[r * 64 + ka + 4]),
                         *(const float4*)(&h[r * 64 + ka + 8]),
                         *(const float4*)(&h[r * 64 + ka + 12]));
        int kb = k0 + half * 16;
        v16bf b0 = *(const v16bf*)(&rT[(tn + r) * 64 + kb]);
        v16bf b1 = *(const v16bf*)(&rT[(tn + 16 + r) * 64 + kb]);
        acc0 = __builtin_amdgcn_wmma_f32_16x16x32_bf16(false, a, false, b0,
                                                       (short)0, acc0, false, false);
        acc1 = __builtin_amdgcn_wmma_f32_16x16x32_bf16(false, a, false, b1,
                                                       (short)0, acc1, false, false);
      }
      #pragma unroll
      for (int v = 0; v < 8; ++v) {
        int m = v + (half << 3);
        hg[m * 192 + tn + r]      = acc0[v] + br[tn + r];
        hg[m * 192 + tn + 16 + r] = acc1[v] + br[tn + 16 + r];
      }
    }
    __syncthreads();
    int tt = dir ? (63 - t) : t;
    #pragma unroll
    for (int u = 0; u < 4; ++u) {
      int idx = u * 256 + tid;
      int row = idx >> 6, j = idx & 63;
      const float* xrow = xg + ((size_t)(b0 + row) * 64 + tt) * 192;
      float xz = xrow[j], xr = xrow[64 + j], xh = xrow[128 + j];
      float hz = hg[row * 192 + j], hr = hg[row * 192 + 64 + j], hh = hg[row * 192 + 128 + j];
      float z  = 1.f / (1.f + __expf(-(xz + hz)));
      float rt = 1.f / (1.f + __expf(-(xr + hr)));
      float c  = tanhf(xh + rt * hh);
      float hn = z * h[idx] + (1.f - z) * c;
      h[idx] = hn;
      ys[((size_t)(b0 + row) * 64 + t) * 64 + j] = hn;
    }
    __syncthreads();
  }
  for (int idx = tid; idx < 16 * 64; idx += 256) {
    int row = idx >> 6, j = idx & 63;
    stb[(((size_t)(b0 + row) * 8 + layer) * 2 + dir) * 64 + j] = h[idx];
  }
}

// final packing: out (B,64,80) = [last | st^T], then out_state = st flat
__global__ void k_fin(const float* __restrict__ lastb, const float* __restrict__ stb,
                      float* __restrict__ outp, long total)
{
  long idx = (long)blockIdx.x * blockDim.x + threadIdx.x;
  if (idx >= total) return;
  const long part1 = (long)BN * 5120;
  if (idx < part1) {
    long b = idx / 5120; int r = (int)(idx % 5120);
    int j = r / 80, e = r % 80;
    outp[idx] = (e < 64) ? lastb[(b << 12) + (j << 6) + e]
                         : stb[((b << 4) + (e - 64)) * 64 + j];
  } else {
    outp[idx] = stb[idx - part1];
  }
}

// ---------------------------------------------------------------------------
static inline void gemm(hipStream_t s, const float* A, const __bf16* Wt,
                        const float* bias, float* C, int M, int N, int K,
                        int rowsA, int rowsC, int rowOff)
{
  int blocks = (M / 128) * (N / 64);
  k_gemm<<<blocks, 256, 0, s>>>(A, Wt, bias, C, M, N, K, rowsA, rowsC, rowOff);
}

extern "C" void kernel_launch(void* const* d_in, const int* in_sizes, int n_in,
                              void* d_out, int out_size, void* d_ws, size_t ws_size,
                              hipStream_t stream)
{
  (void)in_sizes; (void)n_in; (void)out_size; (void)ws_size;
  const float* x     = (const float*)d_in[0];
  const float* state = (const float*)d_in[1];
  const float* cw4   = (const float*)d_in[2];
  const float* cw3   = (const float*)d_in[3];
  const float* cw2   = (const float*)d_in[4];
  const float* cw1   = (const float*)d_in[5];
  const float* cb    = (const float*)d_in[6];
  const float* qkvw  = (const float*)d_in[7];
  const float* qkvb  = (const float*)d_in[8];
  const float* outw  = (const float*)d_in[9];
  const float* outb  = (const float*)d_in[10];
  const float* ffw   = (const float*)d_in[11];
  const float* ang   = (const float*)d_in[12];
  const float* anb   = (const float*)d_in[13];
  const float* gruk  = (const float*)d_in[14];
  const float* grur  = (const float*)d_in[15];
  const float* grub  = (const float*)d_in[16];
  const float* skg   = (const float*)d_in[17];
  const float* skb   = (const float*)d_in[18];

  char* ws = (char*)d_ws;
  size_t off = 0;
  auto alloc = [&](size_t bytes) -> char* {
    char* p = ws + off;
    off += (bytes + 255) & ~(size_t)255;
    return p;
  };

  __bf16* wtc4  = (__bf16*)alloc((size_t)LAY * 4096 * 64 * 2);
  __bf16* wtc3  = (__bf16*)alloc((size_t)LAY * 1728 * 64 * 2);
  __bf16* wtc2  = (__bf16*)alloc((size_t)LAY * 512 * 64 * 2);
  __bf16* wtc1  = (__bf16*)alloc((size_t)LAY * 64 * 64 * 2);
  __bf16* wtqkv = (__bf16*)alloc((size_t)LAY * 2 * 3 * 64 * 64 * 2);
  __bf16* wtout = (__bf16*)alloc((size_t)LAY * 2 * 64 * 64 * 2);
  __bf16* wtgk  = (__bf16*)alloc((size_t)LAY * 2 * 64 * 192 * 2);
  __bf16* wtgr  = (__bf16*)alloc((size_t)LAY * 2 * 64 * 192 * 2);

  float* lastb = (float*)alloc((size_t)BN * 64 * 64 * 4);
  float* mst   = (float*)alloc((size_t)BN * 16 * 64 * 4);
  float* colb  = (float*)alloc((size_t)BN * 27 * 512 * 4);   // covers all im2col
  float* ccb   = (float*)alloc((size_t)BN * 128 * 64 * 4);   // cc (S=100) / ap (S=128)
  float* qb    = (float*)alloc((size_t)BN * 128 * 64 * 4);
  float* kb    = (float*)alloc((size_t)BN * 128 * 64 * 4);
  float* vb    = (float*)alloc((size_t)BN * 128 * 64 * 4);
  float* ob    = (float*)alloc((size_t)BN * 128 * 64 * 4);
  float* mhab  = (float*)alloc((size_t)BN * 128 * 64 * 4);
  float* d0b   = (float*)alloc((size_t)BN * 64 * 64 * 4);
  float* gpb   = (float*)alloc((size_t)BN * 64 * 64 * 4);
  float* xgF   = (float*)alloc((size_t)BN * 64 * 192 * 4);
  float* xgB   = (float*)alloc((size_t)BN * 64 * 192 * 4);
  float* ysF   = (float*)alloc((size_t)BN * 64 * 64 * 4);
  float* ysB   = (float*)alloc((size_t)BN * 64 * 64 * 4);
  float* byp   = (float*)alloc((size_t)LAY * BN * 64 * 64 * 4);
  float* stb   = (float*)alloc((size_t)BN * 16 * 64 * 4);

  // ---- weight prep: convert to bf16 + transpose to (N, K) ----
  auto prep = [&](const float* W, __bf16* Wt, int K, int N, int nm) {
    long total = (long)nm * K * N;
    k_wt<<<(int)((total + 255) / 256), 256, 0, stream>>>(W, Wt, K, N, total);
  };
  prep(cw4, wtc4, 4096, 64, LAY);
  prep(cw3, wtc3, 1728, 64, LAY);
  prep(cw2, wtc2, 512, 64, LAY);
  prep(cw1, wtc1, 64, 64, LAY);
  prep(qkvw, wtqkv, 64, 64, LAY * 2 * 3);
  prep(outw, wtout, 64, 64, LAY * 2);
  prep(gruk, wtgk, 64, 192, LAY * 2);
  prep(grur, wtgr, 64, 192, LAY * 2);

  {
    long total = (long)BN * 64 * 80;
    k_pre<<<(int)((total + 255) / 256), 256, 0, stream>>>(x, state, lastb, mst, total);
  }

  auto mha = [&](const float* in, int S, int i, int a, float* outbuf) {
    for (int m = 0; m < 3; ++m) {
      float* dst = (m == 0) ? qb : (m == 1) ? kb : vb;
      gemm(stream, in, wtqkv + (((size_t)i * 2 + a) * 3 + m) * 4096,
           qkvb + (((size_t)i * 2 + a) * 3 + m) * 64, dst, BN * S, 64, 64, S, S, 0);
    }
    k_attn<<<dim3(BN, 8), 256, 0, stream>>>(qb, kb, vb, ob, S);
    gemm(stream, ob, wtout + ((size_t)i * 2 + a) * 4096,
         outb + ((size_t)i * 2 + a) * 64, outbuf, BN * S, 64, 64, S, S, 0);
  };

  for (int i = 0; i < LAY; ++i) {
    // ---- multi-scale conv -> cc (B,100,64) via im2col + WMMA GEMM ----
    long t4 = (long)BN * 1 * 4096;
    k_im2col<<<(int)((t4 + 255) / 256), 256, 0, stream>>>(lastb, colb, 4, 1, t4);
    gemm(stream, colb, wtc4 + (size_t)i * 4096 * 64, cb + ((size_t)i * 4 + 0) * 64,
         ccb, BN * 1, 64, 4096, 1, 100, 0);
    long t3 = (long)BN * 8 * 1728;
    k_im2col<<<(int)((t3 + 255) / 256), 256, 0, stream>>>(lastb, colb, 3, 2, t3);
    gemm(stream, colb, wtc3 + (size_t)i * 1728 * 64, cb + ((size_t)i * 4 + 1) * 64,
         ccb, BN * 8, 64, 1728, 8, 100, 1);
    long t2 = (long)BN * 27 * 512;
    k_im2col<<<(int)((t2 + 255) / 256), 256, 0, stream>>>(lastb, colb, 2, 3, t2);
    gemm(stream, colb, wtc2 + (size_t)i * 512 * 64, cb + ((size_t)i * 4 + 2) * 64,
         ccb, BN * 27, 64, 512, 27, 100, 9);
    gemm(stream, lastb, wtc1 + (size_t)i * 64 * 64, cb + ((size_t)i * 4 + 3) * 64,
         ccb, BN * 64, 64, 64, 64, 100, 36);

    // ---- MHA #0 (S=100) + FF/LN -> d0 ----
    mha(ccb, 100, i, 0, mhab);
    k_ffln<<<BN, 256, 0, stream>>>(mhab, 100, ffw + ((size_t)i * 2 + 0) * 8192, lastb,
                                   ang + ((size_t)i * 2 + 0) * 64,
                                   anb + ((size_t)i * 2 + 0) * 64, d0b);

    // ---- GRU: x-gates as one big GEMM per dir, fused recurrent kernel ----
    long tt = (long)BN * 4096;
    k_tr<<<(int)((tt + 255) / 256), 256, 0, stream>>>(d0b, gpb, tt);
    gemm(stream, gpb, wtgk + (size_t)(i * 2 + 0) * 12288,
         grub + ((size_t)(i * 2 + 0) * 2 + 0) * 192, xgF, BN * 64, 192, 64, 64, 64, 0);
    gemm(stream, gpb, wtgk + (size_t)(i * 2 + 1) * 12288,
         grub + ((size_t)(i * 2 + 1) * 2 + 0) * 192, xgB, BN * 64, 192, 64, 64, 64, 0);
    k_gru<<<dim3(BN / 16, 2), 256, 0, stream>>>(
        xgF, xgB,
        wtgr + (size_t)(i * 2 + 0) * 12288, wtgr + (size_t)(i * 2 + 1) * 12288,
        grub + ((size_t)(i * 2 + 0) * 2 + 1) * 192,
        grub + ((size_t)(i * 2 + 1) * 2 + 1) * 192,
        mst, i, ysF, ysB, stb);

    // ---- MHA #1 (S=128) on [fore|back] + FF/LN -> bypass[i] ----
    long ta = (long)BN * 8192;
    k_ap<<<(int)((ta + 255) / 256), 256, 0, stream>>>(ysF, ysB, ccb, ta);
    mha(ccb, 128, i, 1, mhab);
    float* bypI = byp + (size_t)i * BN * 4096;
    k_ffln<<<BN, 256, 0, stream>>>(mhab, 128, ffw + ((size_t)i * 2 + 1) * 8192, d0b,
                                   ang + ((size_t)i * 2 + 1) * 64,
                                   anb + ((size_t)i * 2 + 1) * 64, bypI);

    // ---- skip-LN chain ----
    int j = 1, s = 0;
    while (j <= LAY && ((i + 1) % j) == 0 && s < 4) {
      const float* xx = byp + (size_t)(i - j + 1) * BN * 4096;
      const float* yy = (s == 0) ? bypI : lastb;
      k_addln<<<BN * 64, 64, 0, stream>>>(xx, yy, skg + ((size_t)i * 4 + s) * 64,
                                          skb + ((size_t)i * 4 + s) * 64, lastb);
      j <<= 1; ++s;
    }
  }

  {
    long total = (long)BN * 5120 + (long)BN * 1024;
    k_fin<<<(int)((total + 255) / 256), 256, 0, stream>>>(lastb, stb, (float*)d_out, total);
  }
}
